// DecoderLayer_36644660970211
// MI455X (gfx1250) — compile-verified, matrix operand order checked by reading, not run on hardware
//
#include <hip/hip_runtime.h>

// ---------------------------------------------------------------------------
// MS-DeformDETR decoder layer for gfx1250 (MI455X), bf16 WMMA + f32 accum.
// All GEMMs via v_wmma_f32_16x16x32_bf16; self-attention is flash-style
// (online softmax, LDS transpose of P tiles, s_wait_dscnt fences).
// Edge handling: fragment row indices are CLAMPED into bounds (reads stay
// in-bounds, garbage never stored) so the hot loop has no exec-mask branches.
// ---------------------------------------------------------------------------

typedef __bf16 bf16;
typedef __attribute__((ext_vector_type(16))) __bf16 v16bf;
typedef __attribute__((ext_vector_type(2)))  __bf16 bf16x2;
typedef __attribute__((ext_vector_type(8)))  float  v8f;

#define Dc    256
#define NHc   8
#define HDc   32
#define LEVc  3
#define NPc   4
#define DFFc  1024
#define Bc    8
#define LQc   3600
#define LINc  4725

#define WMMA_BF16(a,b,c) \
  __builtin_amdgcn_wmma_f32_16x16x32_bf16(false,(a),false,(b),(short)0,(c),false,false)

// ---- WMMA fragment loaders -------------------------------------------------
// 16-bit A/B layout (ISA 7.12.2): lane L holds row (L&15); half = L>>4 selects
// K interleave:  vgpr r<4  -> K = half*8 + 2r (+0/1)
//                vgpr r>=4 -> K = 16 + half*8 + 2(r-4) (+0/1)
// Compiles to two global_load_b128 (row bytes [0..15] and [32..47]).
__device__ __forceinline__ v16bf wmma_load_frag(const bf16* base, long row, long ld,
                                                long k0, int half) {
  const bf16* p = base + row * ld + k0 + half * 8;
  v16bf a;
#pragma unroll
  for (int r = 0; r < 4; ++r) {
    bf16x2 q = *(const bf16x2*)(p + 2 * r);
    a[2 * r] = q.x; a[2 * r + 1] = q.y;
  }
#pragma unroll
  for (int r = 0; r < 4; ++r) {
    bf16x2 q = *(const bf16x2*)(p + 16 + 2 * r);
    a[8 + 2 * r] = q.x; a[9 + 2 * r] = q.y;
  }
  return a;
}

// Low-K-half only (K 0..15 valid, 16..31 zero) for ragged attention tail.
__device__ __forceinline__ v16bf wmma_load_frag_lo(const bf16* base, long row, long ld,
                                                   long k0, int half) {
  v16bf a = {};
  const bf16* p = base + row * ld + k0 + half * 8;
#pragma unroll
  for (int r = 0; r < 4; ++r) {
    bf16x2 q = *(const bf16x2*)(p + 2 * r);
    a[2 * r] = q.x; a[2 * r + 1] = q.y;
  }
  return a;
}

__device__ __forceinline__ float redmax16(float v) {
  v = fmaxf(v, __shfl_xor(v, 1, 32));
  v = fmaxf(v, __shfl_xor(v, 2, 32));
  v = fmaxf(v, __shfl_xor(v, 4, 32));
  v = fmaxf(v, __shfl_xor(v, 8, 32));
  return v;
}
__device__ __forceinline__ float redsum16(float v) {
  v += __shfl_xor(v, 1, 32);
  v += __shfl_xor(v, 2, 32);
  v += __shfl_xor(v, 4, 32);
  v += __shfl_xor(v, 8, 32);
  return v;
}
__device__ __forceinline__ float redsum32(float v) {
  v = redsum16(v);
  v += __shfl_xor(v, 16, 32);
  return v;
}

// ---- generic bf16 WMMA GEMM:  C[M,N] = act(X[M,K] * W[N,K]^T + bias) -------
// outmode 0: C[m*N+n]
// outmode 1: head-split  ((b*NH+h)*Mi + i)*HD + hd     (m=b*Mi+i, n=h*32+hd)
// outmode 2: head-transp ((b*NH+h)*HD + hd)*Mi + i
__global__ void __launch_bounds__(256)
gemm_wmma_bf16(const bf16* __restrict__ X, const bf16* __restrict__ W,
               const float* __restrict__ bias,
               float* __restrict__ outf, bf16* __restrict__ outb,
               int M, int N, int K, int relu, int outmode, int Mi) {
  const int lane = threadIdx.x & 31;
  const int wave = threadIdx.x >> 5;
  const int half = lane >> 4;
  const int l16  = lane & 15;
  const int m0 = blockIdx.y * 64 + (wave >> 2) * 32;
  const int n0 = blockIdx.x * 256 + (wave & 3) * 64;

  v8f zero8 = {};
  v8f acc[2][4];
#pragma unroll
  for (int i = 0; i < 2; ++i)
#pragma unroll
    for (int j = 0; j < 4; ++j) acc[i][j] = zero8;

  // Clamp read rows into bounds; garbage lanes are masked at store time.
  const int mA0 = min(m0 + l16, M - 1);
  const int mA1 = min(m0 + 16 + l16, M - 1);
  int nB[4];
#pragma unroll
  for (int j = 0; j < 4; ++j) nB[j] = min(n0 + j * 16 + l16, N - 1);

  for (int k0 = 0; k0 < K; k0 += 32) {
    v16bf a0 = wmma_load_frag(X, mA0, K, k0, half);
    v16bf a1 = wmma_load_frag(X, mA1, K, k0, half);
#pragma unroll
    for (int j = 0; j < 4; ++j) {
      v16bf bf = wmma_load_frag(W, nB[j], K, k0, half);
      acc[0][j] = WMMA_BF16(a0, bf, acc[0][j]);
      acc[1][j] = WMMA_BF16(a1, bf, acc[1][j]);
    }
  }

#pragma unroll
  for (int i = 0; i < 2; ++i) {
#pragma unroll
    for (int j = 0; j < 4; ++j) {
      int n = n0 + j * 16 + l16;
      if (n >= N) continue;
      float bv = bias ? bias[n] : 0.f;
#pragma unroll
      for (int r = 0; r < 8; ++r) {
        int m = m0 + i * 16 + r + 8 * half;
        if (m >= M) continue;
        float v = acc[i][j][r] + bv;
        if (relu) v = fmaxf(v, 0.f);
        size_t idx;
        if (outmode == 0) {
          idx = (size_t)m * N + n;
        } else {
          int bb = m / Mi, ii = m - bb * Mi;
          int hh = n >> 5, dd = n & 31;
          if (outmode == 1) idx = ((size_t)(bb * NHc + hh) * Mi + ii) * HDc + dd;
          else              idx = ((size_t)(bb * NHc + hh) * HDc + dd) * (size_t)Mi + ii;
        }
        if (outf) outf[idx] = v;
        if (outb) outb[idx] = (bf16)v;
      }
    }
  }
}

// ---- flash attention: one wave per (b,h, 16-row S tile) --------------------
// Q,K: (b,h,LQ,HD) bf16 row-major.  Vt: (b,h,HD,LQ) bf16.  CTX: (b,LQ,D) bf16.
__global__ void __launch_bounds__(256)
flash_attn(const bf16* __restrict__ Q, const bf16* __restrict__ Kp,
           const bf16* __restrict__ Vt, bf16* __restrict__ CTX) {
  __shared__ bf16 plds[8 * 16 * 32];
  const int lane = threadIdx.x & 31;
  const int wave = threadIdx.x >> 5;
  const int half = lane >> 4;
  const int l16  = lane & 15;

  int wg    = blockIdx.x * 8 + wave;     // [0, B*NH*225)
  int stile = wg % (LQc / 16);
  int bh    = wg / (LQc / 16);
  const int s0 = stile * 16;

  const bf16* Qb = Q  + (size_t)bh * LQc * HDc;
  const bf16* Kb = Kp + (size_t)bh * LQc * HDc;
  const bf16* Vb = Vt + (size_t)bh * HDc * LQc;
  bf16* lds = &plds[wave * 512];

  const float scale = 0.17677669529663689f;   // 1/sqrt(32)
  v16bf qa = wmma_load_frag(Qb, s0 + l16, HDc, 0, half);

  v8f zc = {};
  v8f acc0 = zc, acc1 = zc;
  float mrow[8], lrow[8];
#pragma unroll
  for (int r = 0; r < 8; ++r) { mrow[r] = -3.0e38f; lrow[r] = 0.f; }

  // main loop: 112 blocks of 32 columns (t = 0..3583)
  for (int t0 = 0; t0 < LQc - 16; t0 += 32) {
    v16bf kb0 = wmma_load_frag(Kb, t0 + l16,      HDc, 0, half);
    v16bf kb1 = wmma_load_frag(Kb, t0 + 16 + l16, HDc, 0, half);
    v8f s0v = WMMA_BF16(qa, kb0, zc);
    v8f s1v = WMMA_BF16(qa, kb1, zc);
#pragma unroll
    for (int r = 0; r < 8; ++r) {
      float x0 = s0v[r] * scale, x1 = s1v[r] * scale;
      float mx = redmax16(fmaxf(x0, x1));
      float mn = fmaxf(mrow[r], mx);
      float alpha = __expf(mrow[r] - mn);
      mrow[r] = mn;
      float p0 = __expf(x0 - mn), p1 = __expf(x1 - mn);
      float ps = redsum16(p0 + p1);
      lrow[r] = lrow[r] * alpha + ps;
      acc0[r] *= alpha; acc1[r] *= alpha;
      int row = r + 8 * half;
      lds[row * 32 + l16]      = (bf16)p0;
      lds[row * 32 + 16 + l16] = (bf16)p1;
    }
    asm volatile("s_wait_dscnt 0x0" ::: "memory");
    v16bf pa  = wmma_load_frag(lds, l16, 32, 0, half);
    v16bf vb0 = wmma_load_frag(Vb, l16,      LQc, t0, half);
    v16bf vb1 = wmma_load_frag(Vb, 16 + l16, LQc, t0, half);
    acc0 = WMMA_BF16(pa, vb0, acc0);
    acc1 = WMMA_BF16(pa, vb1, acc1);
  }

  // tail: last 16 columns (t = 3584..3599), upper K half padded with zeros
  {
    const int t0 = LQc - 16;
    v16bf kb0 = wmma_load_frag(Kb, t0 + l16, HDc, 0, half);
    v8f s0v = WMMA_BF16(qa, kb0, zc);
#pragma unroll
    for (int r = 0; r < 8; ++r) {
      float x0 = s0v[r] * scale;
      float mx = redmax16(x0);
      float mn = fmaxf(mrow[r], mx);
      float alpha = __expf(mrow[r] - mn);
      mrow[r] = mn;
      float p0 = __expf(x0 - mn);
      float ps = redsum16(p0);
      lrow[r] = lrow[r] * alpha + ps;
      acc0[r] *= alpha; acc1[r] *= alpha;
      int row = r + 8 * half;
      lds[row * 32 + l16]      = (bf16)p0;
      lds[row * 32 + 16 + l16] = (bf16)0.f;
    }
    asm volatile("s_wait_dscnt 0x0" ::: "memory");
    v16bf pa  = wmma_load_frag(lds, l16, 32, 0, half);
    v16bf vb0 = wmma_load_frag_lo(Vb, l16,      LQc, t0, half);
    v16bf vb1 = wmma_load_frag_lo(Vb, 16 + l16, LQc, t0, half);
    acc0 = WMMA_BF16(pa, vb0, acc0);
    acc1 = WMMA_BF16(pa, vb1, acc1);
  }

  const int bq = bh >> 3, hq = bh & 7;
#pragma unroll
  for (int r = 0; r < 8; ++r) {
    int row = r + 8 * half;
    float inv = 1.0f / lrow[r];
    size_t base = ((size_t)(bq * LQc + s0 + row)) * Dc + hq * HDc;
    CTX[base + l16]      = (bf16)(acc0[r] * inv);
    CTX[base + 16 + l16] = (bf16)(acc1[r] * inv);
  }
}

// ---- LayerNorm fuse: y = LN(xa [+ xb] [+ tgt(s,b,d)]) ; f32 + optional bf16
__global__ void __launch_bounds__(256)
ln_fuse(const float* __restrict__ xa, const float* __restrict__ xb,
        const float* __restrict__ res_sbd,
        const float* __restrict__ g, const float* __restrict__ bt,
        float* __restrict__ outf, int out_sbd,
        bf16* __restrict__ outb, const float* __restrict__ qp) {
  const int lane = threadIdx.x & 31;
  const int wave = threadIdx.x >> 5;
  int row = blockIdx.x * 8 + wave;           // [0, B*LQ)
  int b = row / LQc, s = row - b * LQc;

  float x[8];
#pragma unroll
  for (int j = 0; j < 8; ++j) {
    int d = j * 32 + lane;
    float v = xa[(size_t)row * Dc + d];
    if (xb)      v += xb[(size_t)row * Dc + d];
    if (res_sbd) v += res_sbd[((size_t)s * Bc + b) * Dc + d];
    x[j] = v;
  }
  float sum = 0.f;
#pragma unroll
  for (int j = 0; j < 8; ++j) sum += x[j];
  float mean = redsum32(sum) * (1.f / Dc);
  float var = 0.f;
#pragma unroll
  for (int j = 0; j < 8; ++j) { float dlt = x[j] - mean; var += dlt * dlt; }
  var = redsum32(var) * (1.f / Dc);
  float rinv = rsqrtf(var + 1e-5f);
#pragma unroll
  for (int j = 0; j < 8; ++j) {
    int d = j * 32 + lane;
    float y = (x[j] - mean) * rinv * g[d] + bt[d];
    if (outf) {
      size_t oi = out_sbd ? ((size_t)s * Bc + b) * Dc + d : (size_t)row * Dc + d;
      outf[oi] = y;
    }
    if (outb) outb[(size_t)row * Dc + d] = (bf16)(y + (qp ? qp[d] : 0.f));
  }
}

// ---- deformable sampling: one wave per (b,s,h); lane = hd channel ----------
__global__ void __launch_bounds__(256)
msdeform_sample(const float* __restrict__ VAL, const float* __restrict__ OFF,
                const float* __restrict__ AW, bf16* __restrict__ MSO) {
  const int lane = threadIdx.x & 31;
  const int wave = threadIdx.x >> 5;
  long wg = (long)blockIdx.x * 8 + wave;     // [0, B*LQ*NH)
  int h = (int)(wg % NHc);
  long bs = wg / NHc;
  int s = (int)(bs % LQc);
  int b = (int)(bs / LQc);

  const float* offp = OFF + ((size_t)(b * LQc + s)) * (NHc * LEVc * NPc * 2) + h * (LEVc * NPc * 2);
  const float* awp  = AW  + ((size_t)(b * LQc + s)) * (NHc * LEVc * NPc)     + h * (LEVc * NPc);

  float aw[LEVc * NPc];
  float mx = -3.0e38f;
#pragma unroll
  for (int i = 0; i < LEVc * NPc; ++i) { aw[i] = awp[i]; mx = fmaxf(mx, aw[i]); }
  float ssum = 0.f;
#pragma unroll
  for (int i = 0; i < LEVc * NPc; ++i) { aw[i] = __expf(aw[i] - mx); ssum += aw[i]; }
  float rs = 1.f / ssum;

  float refx = ((s % 60) + 0.5f) * (1.f / 60.f);
  float refy = ((s / 60) + 0.5f) * (1.f / 60.f);

  const int Hls[LEVc] = {60, 30, 15};
  const int Sls[LEVc] = {0, 3600, 4500};
  float acc = 0.f;
#pragma unroll
  for (int l = 0; l < LEVc; ++l) {
    const int Hl = Hls[l], Wl = Hls[l];
    const float* vb = VAL + ((size_t)(b * NHc + h) * LINc + Sls[l]) * HDc;
#pragma unroll
    for (int p = 0; p < NPc; ++p) {
      float a  = aw[l * NPc + p] * rs;
      float ox = offp[(l * NPc + p) * 2 + 0];
      float oy = offp[(l * NPc + p) * 2 + 1];
      float xg = (refx + ox / (float)Wl) * Wl - 0.5f;
      float yg = (refy + oy / (float)Hl) * Hl - 0.5f;
      float x0f = floorf(xg), y0f = floorf(yg);
      float fx = xg - x0f, fy = yg - y0f;
      int x0 = (int)x0f, y0 = (int)y0f;
      float w00 = (1.f - fx) * (1.f - fy) * a;
      float w01 = fx * (1.f - fy) * a;
      float w10 = (1.f - fx) * fy * a;
      float w11 = fx * fy * a;
      bool vx0 = (x0 >= 0) & (x0 < Wl), vx1 = (x0 + 1 >= 0) & (x0 + 1 < Wl);
      bool vy0 = (y0 >= 0) & (y0 < Hl), vy1 = (y0 + 1 >= 0) & (y0 + 1 < Hl);
      if (vx0 & vy0) acc += w00 * vb[((size_t)y0 * Wl + x0) * HDc + lane];
      if (vx1 & vy0) acc += w01 * vb[((size_t)y0 * Wl + x0 + 1) * HDc + lane];
      if (vx0 & vy1) acc += w10 * vb[((size_t)(y0 + 1) * Wl + x0) * HDc + lane];
      if (vx1 & vy1) acc += w11 * vb[((size_t)(y0 + 1) * Wl + x0 + 1) * HDc + lane];
    }
  }
  MSO[((size_t)(b * LQc + s)) * Dc + h * HDc + lane] = (bf16)acc;
}

// ---- small elementwise helpers ---------------------------------------------
__global__ void cvt_f32_bf16(const float* __restrict__ in, bf16* __restrict__ out, long n) {
  for (long i = (long)blockIdx.x * blockDim.x + threadIdx.x; i < n;
       i += (long)gridDim.x * blockDim.x)
    out[i] = (bf16)in[i];
}

// tgt (LQ,B,D) -> XQ = bf16(tgt+qp) (B,LQ,D), XT = bf16(tgt) (B,LQ,D)
__global__ void build_xq_xt(const float* __restrict__ tgt, const float* __restrict__ qp,
                            bf16* __restrict__ XQ, bf16* __restrict__ XT) {
  long idx = (long)blockIdx.x * blockDim.x + threadIdx.x;
  long total = (long)Bc * LQc * Dc;
  if (idx >= total) return;
  int d = (int)(idx % Dc);
  long r = idx / Dc;
  int s = (int)(r % LQc);
  int b = (int)(r / LQc);
  float t = tgt[((size_t)s * Bc + b) * Dc + d];
  XT[idx] = (bf16)t;
  XQ[idx] = (bf16)(t + qp[d]);
}

// ---------------------------------------------------------------------------
extern "C" void kernel_launch(void* const* d_in, const int* in_sizes, int n_in,
                              void* d_out, int out_size, void* d_ws, size_t ws_size,
                              hipStream_t stream) {
  (void)in_sizes; (void)n_in; (void)out_size;
  const float* tgt          = (const float*)d_in[0];
  const float* query_pos    = (const float*)d_in[1];
  const float* src          = (const float*)d_in[2];
  const float* in_proj_w    = (const float*)d_in[5];
  const float* in_proj_b    = (const float*)d_in[6];
  const float* out_proj_w   = (const float*)d_in[7];
  const float* out_proj_b   = (const float*)d_in[8];
  const float* samp_off_w   = (const float*)d_in[9];
  const float* samp_off_b   = (const float*)d_in[10];
  const float* attn_w_w     = (const float*)d_in[11];
  const float* attn_w_b     = (const float*)d_in[12];
  const float* value_proj_w = (const float*)d_in[13];
  const float* value_proj_b = (const float*)d_in[14];
  const float* cross_out_w  = (const float*)d_in[15];
  const float* cross_out_b  = (const float*)d_in[16];
  const float* ln1_g = (const float*)d_in[17];
  const float* ln1_b = (const float*)d_in[18];
  const float* ln2_g = (const float*)d_in[19];
  const float* ln2_b = (const float*)d_in[20];
  const float* ln3_g = (const float*)d_in[21];
  const float* ln3_b = (const float*)d_in[22];
  const float* ffn_w1 = (const float*)d_in[23];
  const float* ffn_b1 = (const float*)d_in[24];
  const float* ffn_w2 = (const float*)d_in[25];
  const float* ffn_b2 = (const float*)d_in[26];

  const size_t NB   = (size_t)Bc * LQc * Dc;     // 7,372,800
  const size_t NSRC = (size_t)Bc * LINc * Dc;    // 9,676,800
  const size_t NWGT = (768 + 256 + 192 + 96 + 256 + 256 + 1024 + 1024) * 256;

  char* ws = (char*)d_ws;
  size_t off = 0;
  auto take = [&](size_t bytes) -> char* {
    char* p = ws + off;
    off = (off + bytes + 255) & ~(size_t)255;
    return p;
  };
  bf16*  WB   = (bf16*)take(NWGT * 2);
  bf16*  XQ   = (bf16*)take(NB * 2);
  bf16*  XT   = (bf16*)take(NB * 2);
  bf16*  SRCB = (bf16*)take(NSRC * 2);
  bf16*  Qh   = (bf16*)take(NB * 2);
  bf16*  Kh   = (bf16*)take(NB * 2);
  bf16*  Vt   = (bf16*)take(NB * 2 + 256);
  float* AOUT = (float*)take(NB * 4);
  float* TF   = (float*)take(NB * 4);
  float* VALF = (float*)take(NSRC * 4);
  float* OFFb = (float*)take((size_t)Bc * LQc * 192 * 4);
  bf16*  FFH  = (bf16*)take((size_t)Bc * LQc * DFFc * 2);
  if (off > ws_size) return;   // workspace too small (needs ~270 MB)

  // phase aliases
  bf16*  CTX = XQ;           // free after QK projections
  float* AW  = (float*)XQ;   // free after attn out-proj consumed CTX
  bf16*  TQP = XT;           // t + query_pos (bf16), free after V projection
  bf16*  MSO = Qh;           // free after attention
  bf16*  TQB = Kh;           // free after attention
  float* XOUT = AOUT;        // free after LN2
  float* FFO  = VALF;        // free after sampling

  bf16* Win    = WB;
  bf16* Wout   = Win   + 768 * 256;
  bf16* Wsamp  = Wout  + 256 * 256;
  bf16* Wattn  = Wsamp + 192 * 256;
  bf16* Wval   = Wattn + 96 * 256;
  bf16* Wcross = Wval  + 256 * 256;
  bf16* Wff1   = Wcross + 256 * 256;
  bf16* Wff2   = Wff1  + 1024 * 256;

  const int M  = Bc * LQc;        // 28800
  const int MV = Bc * LINc;       // 37800
  dim3 blk(256);

  auto cvt = [&](const float* in, bf16* out, long n) {
    int g = (int)((n + 255) / 256); if (g > 4096) g = 4096;
    cvt_f32_bf16<<<g, blk, 0, stream>>>(in, out, n);
  };
  auto gemm = [&](const bf16* X, const bf16* W, const float* bias, float* of, bf16* ob,
                  int Mm, int Nn, int Kk, int relu, int mode, int Mi) {
    dim3 g((Nn + 255) / 256, (Mm + 63) / 64);
    gemm_wmma_bf16<<<g, blk, 0, stream>>>(X, W, bias, of, ob, Mm, Nn, Kk, relu, mode, Mi);
  };

  // 1) weights -> bf16
  cvt(in_proj_w,    Win,    768 * 256);
  cvt(out_proj_w,   Wout,   256 * 256);
  cvt(samp_off_w,   Wsamp,  192 * 256);
  cvt(attn_w_w,     Wattn,  96 * 256);
  cvt(value_proj_w, Wval,   256 * 256);
  cvt(cross_out_w,  Wcross, 256 * 256);
  cvt(ffn_w1,       Wff1,   1024 * 256);
  cvt(ffn_w2,       Wff2,   256 * 1024);

  // 2) activations -> bf16 (with (LQ,B)->(B,LQ) transpose for tgt)
  build_xq_xt<<<(int)((NB + 255) / 256), blk, 0, stream>>>(tgt, query_pos, XQ, XT);
  cvt(src, SRCB, (long)NSRC);

  // 3) QKV projections straight into head layouts
  gemm(XQ, Win,              in_proj_b,       nullptr, Qh, M, Dc, Dc, 0, 1, LQc);
  gemm(XQ, Win + 256 * 256,  in_proj_b + 256, nullptr, Kh, M, Dc, Dc, 0, 1, LQc);
  gemm(XT, Win + 512 * 256,  in_proj_b + 512, nullptr, Vt, M, Dc, Dc, 0, 2, LQc);

  // 4) self-attention (flash) -> CTX bf16 (B,LQ,D)
  flash_attn<<<(Bc * NHc * (LQc / 16)) / 8, blk, 0, stream>>>(Qh, Kh, Vt, CTX);

  // 5) out-proj, +tgt residual, LN2 -> t (f32) and (t+query_pos) bf16
  gemm(CTX, Wout, out_proj_b, AOUT, nullptr, M, Dc, Dc, 0, 0, 0);
  ln_fuse<<<M / 8, blk, 0, stream>>>(AOUT, nullptr, tgt, ln2_g, ln2_b,
                                     TF, 0, TQP, query_pos);

  // 6) deformable attention
  gemm(TQP,  Wsamp, samp_off_b,   OFFb, nullptr, M, 192, Dc, 0, 0, 0);
  gemm(TQP,  Wattn, attn_w_b,     AW,   nullptr, M, 96,  Dc, 0, 0, 0);
  gemm(SRCB, Wval,  value_proj_b, VALF, nullptr, MV, Dc, Dc, 0, 1, LINc);
  msdeform_sample<<<(Bc * LQc * NHc) / 8, blk, 0, stream>>>(VALF, OFFb, AW, MSO);
  gemm(MSO, Wcross, cross_out_b, XOUT, nullptr, M, Dc, Dc, 0, 0, 0);
  ln_fuse<<<M / 8, blk, 0, stream>>>(XOUT, TF, nullptr, ln1_g, ln1_b,
                                     TF, 0, TQB, nullptr);

  // 7) FFN + LN3, write d_out in (LQ,B,D) order
  gemm(TQB, Wff1, ffn_b1, nullptr, FFH, M, DFFc, Dc, 1, 0, 0);
  gemm(FFH, Wff2, ffn_b2, FFO, nullptr, M, Dc, DFFc, 0, 0, 0);
  ln_fuse<<<M / 8, blk, 0, stream>>>(FFO, TF, nullptr, ln3_g, ln3_b,
                                     (float*)d_out, 1, nullptr, nullptr);
}
